// AlternatingLSTM_49503793054277
// MI455X (gfx1250) — matrix operand order repaired and checked by reading, not compiled, not run on hardware
//
#include <hip/hip_runtime.h>
#include <math.h>

// ---------------- problem constants ----------------
#define B_ 32
#define T_ 512
#define D_ 512
#define H_ 512
#define L_ 8
#define G6 (6 * H_)      // 3072
#define G5 (5 * H_)      // 2560
#define MROWS (B_ * T_)  // 16384
#define KB 128           // K staging width for the projection GEMM
#define APAD 8           // bf16 elements of LDS row padding (16B, one bank-group step)

typedef __attribute__((ext_vector_type(16))) __bf16  v16bf;
typedef __attribute__((ext_vector_type(8)))  float   v8f;
typedef __attribute__((ext_vector_type(4)))  unsigned u32x4;
typedef __attribute__((ext_vector_type(8)))  int      i32x8;
typedef __attribute__((ext_vector_type(4)))  int      i32x4;

__device__ __forceinline__ float sigm_(float x) { return 1.0f / (1.0f + __expf(-x)); }

// Tensor Data Mover: issue a TDM load described by D# groups 0/1 (<=2D tensor).
// ROCm 7.2 (clang-22) builtin takes 5 args; amdgpu-toolchain (clang-23) takes 6.
__device__ __forceinline__ void tdm_load_to_lds(u32x4 g0, i32x8 g1) {
    i32x4 z4 = {0, 0, 0, 0};
#if __clang_major__ >= 23
    i32x8 z8 = {0, 0, 0, 0, 0, 0, 0, 0};
    __builtin_amdgcn_tensor_load_to_lds(g0, g1, z4, z4, z8, 0);
#else
    __builtin_amdgcn_tensor_load_to_lds(g0, g1, z4, z4, 0);
#endif
}

// ---------------- elementwise helpers ----------------
__global__ void alstm_cvt_f32_bf16(const float* __restrict__ s, __bf16* __restrict__ d, int n) {
    int i = blockIdx.x * blockDim.x + threadIdx.x;
    if (i < n) d[i] = (__bf16)s[i];
}

__global__ void alstm_zero_u32(unsigned* __restrict__ p, int n) {
    int i = blockIdx.x * blockDim.x + threadIdx.x;
    if (i < n) p[i] = 0u;
}

// ---------------- input-projection GEMM: C[M,N] = A[M,K] x W[N,K]^T ----------------
// block = 256 threads = 8 waves; block tile = 32 rows x 256 cols.
// Each wave owns 4 accumulators (16x64). A (32 x 128 bf16) tiles are DMA'd into
// double-buffered LDS by the Tensor Data Mover with 16B/row padding (bank-conflict-free),
// overlapped with WMMA compute on the other buffer.
__global__ __launch_bounds__(256) void alstm_gemm_xproj(
    const __bf16* __restrict__ A, const __bf16* __restrict__ W,
    float* __restrict__ C, int M, int N, int K)
{
    __shared__ __bf16 As[2][32 * (KB + APAD)];   // 2 x 8.5 KB

    const int tid   = threadIdx.x;
    const int wave  = tid >> 5;
    const int lane  = tid & 31;
    const int rt    = wave >> 2;                 // 0..1 row tile
    const int ct    = wave & 3;                  // 0..3 col quad
    const int row0  = blockIdx.y * 32;
    const int colb  = blockIdx.x * 256 + ct * 64;
    const int m     = lane & 15;
    const int half  = lane >> 4;

    // D# group1 for a 32 x KB bf16 tile of the [M,K] tensor, with LDS padding:
    // data_size=1 (2B); pad_enable; pad_interval=5 (256B); pad_amount=3 (16B).
    const int g1w0 = 0x00010000 | (1 << 20) | (5 << 22) | (3 << 25);
    const i32x8 g1 = { g1w0,
                       (K & 0xFFFF) << 16,                          // tensor_dim0[15:0]
                       ((K >> 16) & 0xFFFF) | ((M & 0xFFFF) << 16), // dim0 hi | tensor_dim1 lo
                       ((M >> 16) & 0xFFFF) | (KB << 16),           // dim1 hi | tile_dim0
                       32,                                          // tile_dim1
                       K, 0, 0 };                                   // tensor_dim0_stride

    v8f acc[4] = { {}, {}, {}, {} };

    // prologue: DMA first A tile into buffer 0
    if (wave == 0) {
        unsigned lds = (unsigned)(size_t)(void*)&As[0][0];
        unsigned long long ga = (unsigned long long)(size_t)(A + (size_t)row0 * K);
        u32x4 g0 = { 1u, lds, (unsigned)ga, (unsigned)(ga >> 32) | 0x80000000u };
        tdm_load_to_lds(g0, g1);
        __builtin_amdgcn_s_wait_tensorcnt(0);
    }
    __syncthreads();

    int cur = 0;
    for (int kb = 0; kb < K; kb += KB) {
        // overlap: DMA next tile into the other buffer while we compute on this one
        if (wave == 0 && kb + KB < K) {
            unsigned lds = (unsigned)(size_t)(void*)&As[cur ^ 1][0];
            unsigned long long ga =
                (unsigned long long)(size_t)(A + (size_t)row0 * K + kb + KB);
            u32x4 g0 = { 1u, lds, (unsigned)ga, (unsigned)(ga >> 32) | 0x80000000u };
            tdm_load_to_lds(g0, g1);
        }
        __builtin_prefetch(W + (size_t)(colb + m) * K + kb + KB, 0, 1);

#pragma unroll
        for (int kk = 0; kk < KB; kk += 32) {
            v16bf a;
#pragma unroll
            for (int i = 0; i < 16; ++i) {
                int kl = kk + half * 16 + i;
                a[i] = As[cur][(rt * 16 + m) * (KB + APAD) + kl];
            }
#pragma unroll
            for (int j = 0; j < 4; ++j) {
                const __bf16* wsrc = W + (size_t)(colb + j * 16 + m) * K + kb + kk + half * 16;
                v16bf b;
#pragma unroll
                for (int i = 0; i < 16; ++i) b[i] = wsrc[i];
                acc[j] = __builtin_amdgcn_wmma_f32_16x16x32_bf16(
                    false, a, false, b, (short)0, acc[j], false, false);
            }
        }
        if (wave == 0) __builtin_amdgcn_s_wait_tensorcnt(0);
        __syncthreads();
        cur ^= 1;
    }

#pragma unroll
    for (int j = 0; j < 4; ++j) {
        float* cp = C + (size_t)(row0 + rt * 16 + half * 8) * N + colb + j * 16 + m;
#pragma unroll
        for (int i = 0; i < 8; ++i) cp[(size_t)i * N] = acc[j][i];
    }
}

// ---------------- one recurrence time-step ----------------
// grid = 32 blocks (one 16-column group each), block = 320 threads = 10 waves.
// h (32x512 bf16) is DMA'd into LDS by the TDM with 16B/row padding; wave
// (gate 0..4, rtile 0..1) computes one 16x16 tile of hg = h x Wh_slice^T;
// then threads 0..255 fuse gates, cell update, highway mix, masking, stores.
__global__ __launch_bounds__(320) void alstm_step(
    const float*  __restrict__ xg,       // [B*T, 6H] fp32 pre-activations
    const __bf16* __restrict__ Wh,       // [5H, H] bf16 (layer slice)
    const float*  __restrict__ bh,       // [5H] fp32 (layer slice)
    const __bf16* __restrict__ h_in,     // [B, H] bf16
    __bf16*       __restrict__ h_out,    // [B, H] bf16
    float*        __restrict__ c,        // [B, H] fp32
    const int*    __restrict__ lengths,  // [B]
    float*        __restrict__ out,      // [B, T, H] fp32
    int t_act)
{
    __shared__ __bf16 hs[B_ * (H_ + APAD)];   // padded h state (33 KB)
    __shared__ float  hg[5][B_][16];          // gate pre-acts (10 KB)

    const int tid = threadIdx.x;

    // DMA h into LDS: 1D tile of 4096 x 8B, pad 16B (pad_amount=3) every
    // 1024B (pad_interval=7) -> LDS row stride H_+APAD bf16.
    if (tid < 32) {
        unsigned lds = (unsigned)(size_t)(void*)hs;
        unsigned long long ga = (unsigned long long)(size_t)h_in;
        u32x4 g0 = { 1u, lds, (unsigned)ga, (unsigned)(ga >> 32) | 0x80000000u };
        const int g1w0 = 0x00030000 | (1 << 20) | (7 << 22) | (3 << 25);
        i32x8 g1 = { g1w0,
                     0x10000000,   // tensor_dim0[15:0]=4096 in [31:16]
                     0x00010000,   // tensor_dim1=1 in [31:16]
                     0x10000000,   // tile_dim0=4096 in [31:16]
                     1,            // tile_dim1
                     4096, 0, 0 }; // tensor_dim0_stride
        tdm_load_to_lds(g0, g1);
        __builtin_amdgcn_s_wait_tensorcnt(0);
    }
    __syncthreads();

    const int wave  = tid >> 5;
    const int lane  = tid & 31;
    const int gate  = wave % 5;
    const int rtile = wave / 5;          // 0..1
    const int j0    = blockIdx.x * 16;
    const int m     = lane & 15;
    const int half  = lane >> 4;

    v8f acc = {};
    const __bf16* wrow = Wh + (size_t)(gate * H_ + j0 + m) * H_;
#pragma unroll 4
    for (int k0 = 0; k0 < H_; k0 += 32) {
        v16bf a, b;
#pragma unroll
        for (int i = 0; i < 16; ++i) {
            int k = k0 + half * 16 + i;
            a[i] = hs[(rtile * 16 + m) * (H_ + APAD) + k];
        }
#pragma unroll
        for (int i = 0; i < 16; ++i) b[i] = wrow[k0 + half * 16 + i];
        acc = __builtin_amdgcn_wmma_f32_16x16x32_bf16(
            false, a, false, b, (short)0, acc, false, false);
    }

    const float bias = bh[gate * H_ + j0 + m];
#pragma unroll
    for (int i = 0; i < 8; ++i)
        hg[gate][rtile * 16 + half * 8 + i][m] = acc[i] + bias;
    __syncthreads();

    if (tid < 256) {
#pragma unroll
        for (int e = tid; e < B_ * 16; e += 256) {
            const int bb = e >> 4;
            const int jj = e & 15;
            const int n  = j0 + jj;
            const float* xr = xg + ((size_t)bb * T_ + t_act) * G6;

            float ig = sigm_(xr[0 * H_ + n] + hg[0][bb][jj]);
            float fg = sigm_(xr[1 * H_ + n] + hg[1][bb][jj]);
            float gg = tanhf(xr[2 * H_ + n] + hg[2][bb][jj]);
            float og = sigm_(xr[3 * H_ + n] + hg[3][bb][jj]);
            float rg = sigm_(xr[4 * H_ + n] + hg[4][bb][jj]);

            float c_old = c[bb * H_ + n];
            float c_new = ig * gg + fg * c_old;
            float h_pr  = og * tanhf(c_new);
            float h_hat = rg * h_pr + (1.0f - rg) * xr[5 * H_ + n];

            bool v = t_act < lengths[bb];
            c_new = v ? c_new : 0.0f;
            float h_new = v ? h_hat : 0.0f;

            c[bb * H_ + n]     = c_new;
            h_out[bb * H_ + n] = (__bf16)h_new;
            out[((size_t)bb * T_ + t_act) * H_ + n] = h_new;
        }
    }
}

// ---------------- host driver ----------------
extern "C" void kernel_launch(void* const* d_in, const int* in_sizes, int n_in,
                              void* d_out, int out_size, void* d_ws, size_t ws_size,
                              hipStream_t stream) {
    (void)in_sizes; (void)n_in; (void)out_size; (void)ws_size;

    const float* x       = (const float*)d_in[0];
    const int*   lengths = (const int*)  d_in[1];
    const float* Wx      = (const float*)d_in[2];
    const float* Wh      = (const float*)d_in[3];
    const float* bh      = (const float*)d_in[4];
    float*       outp    = (float*)d_out;

    // workspace carve (all sizes multiples of 256B)
    char*  ws  = (char*)d_ws;
    size_t off = 0;
    auto carve = [&](size_t bytes) { void* p = ws + off; off += (bytes + 255) & ~(size_t)255; return p; };

    float*  xgbuf = (float*) carve(sizeof(float)  * (size_t)MROWS * G6);   // 201 MB
    float*  bufA  = (float*) carve(sizeof(float)  * (size_t)MROWS * H_);   //  32 MB
    float*  bufB  = (float*) carve(sizeof(float)  * (size_t)MROWS * H_);   //  32 MB
    float*  cbuf  = (float*) carve(sizeof(float)  * B_ * H_);              //  64 KB  } zeroed
    __bf16* hA    = (__bf16*)carve(sizeof(__bf16) * B_ * H_);              //  32 KB  } together
    __bf16* hB    = (__bf16*)carve(sizeof(__bf16) * B_ * H_);              //  32 KB  }
    __bf16* Wxb   = (__bf16*)carve(sizeof(__bf16) * (size_t)L_ * G6 * D_);
    __bf16* Whb   = (__bf16*)carve(sizeof(__bf16) * (size_t)L_ * G5 * H_);
    __bf16* curb  = (__bf16*)carve(sizeof(__bf16) * (size_t)MROWS * D_);

    // convert weights to bf16 once per call
    {
        int n1 = L_ * G6 * D_;
        alstm_cvt_f32_bf16<<<(n1 + 255) / 256, 256, 0, stream>>>(Wx, Wxb, n1);
        int n2 = L_ * G5 * H_;
        alstm_cvt_f32_bf16<<<(n2 + 255) / 256, 256, 0, stream>>>(Wh, Whb, n2);
    }

    const float* cur = x;
    for (int l = 0; l < L_; ++l) {
        // layer input -> bf16
        int nc = MROWS * D_;
        alstm_cvt_f32_bf16<<<(nc + 255) / 256, 256, 0, stream>>>(cur, curb, nc);

        // x2g = cur x Wx[l]^T    (16384 x 3072, K = 512)
        dim3 ggrid(G6 / 256, MROWS / 32);
        alstm_gemm_xproj<<<ggrid, 256, 0, stream>>>(
            curb, Wxb + (size_t)l * G6 * D_, xgbuf, MROWS, G6, D_);

        // zero c, hA, hB (contiguous region: 64KB + 32KB + 32KB = 32768 u32)
        alstm_zero_u32<<<128, 256, 0, stream>>>((unsigned*)cbuf, 32768);

        const bool rev = (l & 1) != 0;
        float* lout = (l == L_ - 1) ? outp : ((l & 1) ? bufB : bufA);

        for (int t = 0; t < T_; ++t) {
            int t_act = rev ? (T_ - 1 - t) : t;
            const __bf16* hin  = (t & 1) ? hB : hA;
            __bf16*       hout = (t & 1) ? hA : hB;
            alstm_step<<<H_ / 16, 320, 0, stream>>>(
                xgbuf, Whb + (size_t)l * G5 * H_, bh + (size_t)l * G5,
                hin, hout, cbuf, lengths, lout, t_act);
        }
        cur = lout;
    }
}
